// GCN_LSTM_14139032338890
// MI455X (gfx1250) — matrix operand order repaired and checked by reading, not compiled, not run on hardware
//
#include <hip/hip_runtime.h>
#include <hip/hip_bf16.h>

// ---------------------------------------------------------------------------
// Types for WMMA
// ---------------------------------------------------------------------------
typedef __attribute__((ext_vector_type(16))) __bf16        v16bf;
typedef __attribute__((ext_vector_type(8)))  float         v8f;
typedef __attribute__((ext_vector_type(8)))  unsigned int  v8u;

#define N_NODES 8192
#define N_EDGES 262144
#define D_IN    128
#define HDIM    512
#define G4      2048   // 4*H
#define D_OUT   64

#define LDA_PAD 40     // padded row length (elements) for LDS tiles

static __device__ __forceinline__ unsigned short f2bf(float f) {
    unsigned int u = __builtin_bit_cast(unsigned int, f);
    u = (u + 0x7FFFu + ((u >> 16) & 1u)) >> 16;   // round-to-nearest-even
    return (unsigned short)u;
}

// ---------------------------------------------------------------------------
// Elementwise conversion / transpose kernels
// ---------------------------------------------------------------------------
__global__ void k_cvt_bf16(const float* __restrict__ in,
                           unsigned short* __restrict__ out, int n) {
    int i = blockIdx.x * blockDim.x + threadIdx.x;
    if (i < n) out[i] = f2bf(in[i]);
}

// in: [K][N] fp32 row-major  ->  out: [N][K] bf16 row-major  (B-transposed)
__global__ void k_cvt_bf16_T(const float* __restrict__ in,
                             unsigned short* __restrict__ out, int K, int N) {
    int i = blockIdx.x * blockDim.x + threadIdx.x;
    if (i < K * N) {
        int k = i / N, n = i - k * N;
        out[(size_t)n * K + k] = f2bf(in[i]);
    }
}

// in: [R][C] fp32 -> out: [C][R] fp32   (for Whh -> WhhT)
__global__ void k_transpose_f32(const float* __restrict__ in,
                                float* __restrict__ out, int R, int C) {
    int i = blockIdx.x * blockDim.x + threadIdx.x;
    if (i < R * C) {
        int r = i / C, c = i - r * C;
        out[(size_t)c * R + r] = in[i];
    }
}

// ---------------------------------------------------------------------------
// WMMA bf16 GEMM with LDS staging:  C[M,N] = A[M,K]*B[K,N] (+bias0+bias1)(+ReLU)
//   A  : bf16 row-major [M][K];  Bt : bf16 row-major [N][K]  (B transposed)
// Block = 256 threads = 8 waves; block tile 128x64.
//   Per k-step: stage A(128x32)=8KB + B(64x32)=4KB into LDS, all waves reuse.
//   Wave w: rows (w&3)*32 (2 row tiles), cols (w>>2)*32 (2 col tiles)
//           -> 4 accumulators, 4 WMMAs per k-step, each fragment reused 2x.
// Global-side arithmetic intensity ~43 FLOP/byte (vs 16 without staging).
// ---------------------------------------------------------------------------
__global__ __launch_bounds__(256)
void k_gemm_bf16_wmma(const unsigned short* __restrict__ A,
                      const unsigned short* __restrict__ Bt,
                      float* __restrict__ C,
                      const float* __restrict__ bias0,
                      const float* __restrict__ bias1,
                      int M, int N, int K, int relu) {
    __shared__ unsigned short a_lds[128 * LDA_PAD];   // 10 KB
    __shared__ unsigned short b_lds[64  * LDA_PAD];   // 5 KB

    const int tid  = threadIdx.x;
    const int wave = tid >> 5;
    const int lane = tid & 31;
    const int ln   = lane & 15;       // 0..15
    const int lh   = lane >> 4;       // 0..1

    const int m0b = blockIdx.x * 128;
    const int n0b = blockIdx.y * 64;
    const int rm  = (wave & 3) * 32;  // wave row base within block tile
    const int cn  = (wave >> 2) * 32; // wave col base within block tile

    // staging assignments (per thread)
    const int ar0 = tid >> 2;                 // A chunk 0: row 0..63
    const int ar1 = (tid + 256) >> 2;         // A chunk 1: row 64..127
    const int ako = (tid & 3) * 8;            // k-offset in elements (16B chunk)
    const int bcol = tid >> 2;                // B: col 0..63
    const unsigned short* Ar0 = A  + (size_t)(m0b + ar0) * K + ako;
    const unsigned short* Ar1 = A  + (size_t)(m0b + ar1) * K + ako;
    const unsigned short* Bc  = Bt + (size_t)(n0b + bcol) * K + ako;

    const int kb = lh * 16;   // A-fragment K-half select (contiguous, matches B)
    const int kh = lh * 16;   // B-fragment K-half select (contiguous, matches A)

    v8f acc00 = {}, acc01 = {}, acc10 = {}, acc11 = {};

    for (int k0 = 0; k0 < K; k0 += 32) {
        __syncthreads();   // previous compute done before overwriting tiles

        // ---- stage A (128x32) and B (64x32) slabs into LDS ----
        {
            const uint4 va0 = *(const uint4*)(Ar0 + k0);
            const uint4 va1 = *(const uint4*)(Ar1 + k0);
            const uint4 vb  = *(const uint4*)(Bc  + k0);
            __builtin_prefetch(Ar0 + k0 + 32, 0, 1);
            __builtin_prefetch(Ar1 + k0 + 32, 0, 1);
            __builtin_prefetch(Bc  + k0 + 32, 0, 1);
            *(uint2*)&a_lds[ar0 * LDA_PAD + ako]     = make_uint2(va0.x, va0.y);
            *(uint2*)&a_lds[ar0 * LDA_PAD + ako + 4] = make_uint2(va0.z, va0.w);
            *(uint2*)&a_lds[ar1 * LDA_PAD + ako]     = make_uint2(va1.x, va1.y);
            *(uint2*)&a_lds[ar1 * LDA_PAD + ako + 4] = make_uint2(va1.z, va1.w);
            *(uint2*)&b_lds[bcol * LDA_PAD + ako]     = make_uint2(vb.x, vb.y);
            *(uint2*)&b_lds[bcol * LDA_PAD + ako + 4] = make_uint2(vb.z, vb.w);
        }
        __syncthreads();

        // ---- build fragments from LDS (contiguous k = 16*lh + s for A and B) ----
        v8u au0, au1, bu0, bu1;
        const int am0 = (rm + ln)      * LDA_PAD;
        const int am1 = (rm + 16 + ln) * LDA_PAD;
        const int bn0 = (cn + ln)      * LDA_PAD;
        const int bn1 = (cn + 16 + ln) * LDA_PAD;
        #pragma unroll
        for (int v = 0; v < 4; ++v) {
            au0[v] = *(const unsigned int*)&a_lds[am0 + kb + 2 * v];
            au1[v] = *(const unsigned int*)&a_lds[am1 + kb + 2 * v];
        }
        #pragma unroll
        for (int v = 4; v < 8; ++v) {
            au0[v] = *(const unsigned int*)&a_lds[am0 + kb + 2 * v];
            au1[v] = *(const unsigned int*)&a_lds[am1 + kb + 2 * v];
        }
        #pragma unroll
        for (int v = 0; v < 8; ++v) {
            bu0[v] = *(const unsigned int*)&b_lds[bn0 + kh + 2 * v];
            bu1[v] = *(const unsigned int*)&b_lds[bn1 + kh + 2 * v];
        }

        const v16bf a0 = __builtin_bit_cast(v16bf, au0);
        const v16bf a1 = __builtin_bit_cast(v16bf, au1);
        const v16bf b0 = __builtin_bit_cast(v16bf, bu0);
        const v16bf b1 = __builtin_bit_cast(v16bf, bu1);

        acc00 = __builtin_amdgcn_wmma_f32_16x16x32_bf16(false, a0, false, b0, (short)0, acc00, false, false);
        acc01 = __builtin_amdgcn_wmma_f32_16x16x32_bf16(false, a0, false, b1, (short)0, acc01, false, false);
        acc10 = __builtin_amdgcn_wmma_f32_16x16x32_bf16(false, a1, false, b0, (short)0, acc10, false, false);
        acc11 = __builtin_amdgcn_wmma_f32_16x16x32_bf16(false, a1, false, b1, (short)0, acc11, false, false);
    }

    // ---- epilogue: bias + optional relu, store (C/D layout: lane=N, vgpr=M) --
    const int cn0 = n0b + cn + ln;
    const int cn1 = n0b + cn + 16 + ln;
    float bv0 = 0.f, bv1 = 0.f;
    if (bias0) { bv0 += bias0[cn0]; bv1 += bias0[cn1]; }
    if (bias1) { bv0 += bias1[cn0]; bv1 += bias1[cn1]; }

    #pragma unroll
    for (int v = 0; v < 8; ++v) {
        const int mA = m0b + rm + lh * 8 + v;        // row tile 0
        const int mB = mA + 16;                      // row tile 1
        float x00 = acc00[v] + bv0, x01 = acc01[v] + bv1;
        float x10 = acc10[v] + bv0, x11 = acc11[v] + bv1;
        if (relu) {
            x00 = fmaxf(x00, 0.f); x01 = fmaxf(x01, 0.f);
            x10 = fmaxf(x10, 0.f); x11 = fmaxf(x11, 0.f);
        }
        C[(size_t)mA * N + cn0] = x00;
        C[(size_t)mA * N + cn1] = x01;
        C[(size_t)mB * N + cn0] = x10;
        C[(size_t)mB * N + cn1] = x11;
    }
}

// ---------------------------------------------------------------------------
// Graph degree / normalization
// ---------------------------------------------------------------------------
__global__ void k_deg_init(float* __restrict__ deg, int n) {
    int i = blockIdx.x * blockDim.x + threadIdx.x;
    if (i < n) deg[i] = 1.0f;                   // self-loop contribution
}
__global__ void k_deg_edges(const int* __restrict__ dst,
                            float* __restrict__ deg, int e) {
    int i = blockIdx.x * blockDim.x + threadIdx.x;
    if (i < e) atomicAdd(&deg[dst[i]], 1.0f);
}
__global__ void k_dinv(float* __restrict__ deg, int n) {
    int i = blockIdx.x * blockDim.x + threadIdx.x;
    if (i < n) deg[i] = rsqrtf(deg[i]);         // deg >= 1 always
}

// out[n][h] = dinv[n]^2 * xw[n][h] + b[h]   (self-loop term + bias)
__global__ void k_agg_init(const float* __restrict__ xw,
                           const float* __restrict__ dinv,
                           const float* __restrict__ b,
                           float* __restrict__ out) {
    int i = blockIdx.x * blockDim.x + threadIdx.x;
    if (i < N_NODES * HDIM) {
        int n = i >> 9, h = i & (HDIM - 1);
        float dn = dinv[n];
        out[i] = dn * dn * xw[i] + b[h];
    }
}

// out[d] += dinv[s]*dinv[d] * xw[s]  : one thread per (edge, 4-feature chunk)
__global__ void k_scatter(const int* __restrict__ src,
                          const int* __restrict__ dst,
                          const float* __restrict__ dinv,
                          const float* __restrict__ xw,
                          float* __restrict__ out) {
    long long i = (long long)blockIdx.x * blockDim.x + threadIdx.x;
    const long long total = (long long)N_EDGES * (HDIM / 4);
    if (i >= total) return;
    int e  = (int)(i / (HDIM / 4));
    int f  = ((int)(i % (HDIM / 4))) * 4;
    int s  = src[e], d = dst[e];
    float nrm = dinv[s] * dinv[d];
    const float4 v = *(const float4*)(xw + (size_t)s * HDIM + f);
    float* o = out + (size_t)d * HDIM + f;
    atomicAdd(o + 0, nrm * v.x);
    atomicAdd(o + 1, nrm * v.y);
    atomicAdd(o + 2, nrm * v.z);
    atomicAdd(o + 3, nrm * v.w);
}

__global__ void k_relu(float* __restrict__ x, int n) {
    int i = blockIdx.x * blockDim.x + threadIdx.x;
    if (i < n) x[i] = fmaxf(x[i], 0.f);
}

// ---------------------------------------------------------------------------
// LSTM sequential scan (single workgroup, h resident in LDS, Whh^T coalesced)
//   xp   : [T][2048]  precomputed x@Wih.T + (bih+bhh)
//   whhT : [512][2048] (Whh transposed)
//   hs   : [T][512] output hidden states
// Thread j owns feature j (gates j, 512+j, 1024+j, 1536+j).
// ---------------------------------------------------------------------------
__global__ __launch_bounds__(512)
void k_lstm_scan(const float* __restrict__ xp,
                 const float* __restrict__ whhT,
                 float* __restrict__ hs, int T) {
    __shared__ float hsm[HDIM];
    const int j = threadIdx.x;
    hsm[j] = 0.f;
    float c = 0.f;
    __syncthreads();

    for (int t = 0; t < T; ++t) {
        const float* xpt = xp + (size_t)t * G4;
        float gi = xpt[j];
        float gf = xpt[HDIM + j];
        float gg = xpt[2 * HDIM + j];
        float go = xpt[3 * HDIM + j];

        for (int k = 0; k < HDIM; ++k) {
            const float hk = hsm[k];                    // LDS broadcast
            const float* w = whhT + (size_t)k * G4;     // coalesced across j
            gi = fmaf(w[j],            hk, gi);
            gf = fmaf(w[HDIM + j],     hk, gf);
            gg = fmaf(w[2 * HDIM + j], hk, gg);
            go = fmaf(w[3 * HDIM + j], hk, go);
        }

        const float i_ = 1.f / (1.f + __expf(-gi));
        const float f_ = 1.f / (1.f + __expf(-gf));
        const float o_ = 1.f / (1.f + __expf(-go));
        c = f_ * c + i_ * tanhf(gg);
        const float hn = o_ * tanhf(c);

        __syncthreads();          // all reads of old h done
        hsm[j] = hn;
        hs[(size_t)t * HDIM + j] = hn;
        __syncthreads();          // new h visible before next step
    }
}

// ---------------------------------------------------------------------------
// Launch
// ---------------------------------------------------------------------------
static inline int cdiv(long long a, long long b) { return (int)((a + b - 1) / b); }
#define ALIGN256(x) (((x) + 255) & ~(size_t)255)

extern "C" void kernel_launch(void* const* d_in, const int* in_sizes, int n_in,
                              void* d_out, int out_size, void* d_ws, size_t ws_size,
                              hipStream_t stream) {
    const float* x        = (const float*)d_in[0];
    const int*   ei       = (const int*)  d_in[1];
    const float* gcn_w[3] = { (const float*)d_in[2], (const float*)d_in[4], (const float*)d_in[6] };
    const float* gcn_b[3] = { (const float*)d_in[3], (const float*)d_in[5], (const float*)d_in[7] };
    const float* wih0 = (const float*)d_in[8];
    const float* whh0 = (const float*)d_in[9];
    const float* bih0 = (const float*)d_in[10];
    const float* bhh0 = (const float*)d_in[11];
    const float* wih1 = (const float*)d_in[12];
    const float* whh1 = (const float*)d_in[13];
    const float* bih1 = (const float*)d_in[14];
    const float* bhh1 = (const float*)d_in[15];
    const float* lin_w = (const float*)d_in[16];
    const float* lin_b = (const float*)d_in[17];
    float* out = (float*)d_out;

    const int* src = ei;
    const int* dst = ei + N_EDGES;

    // ---- workspace carve-up -------------------------------------------------
    char* ws = (char*)d_ws;
    size_t off = 0;
    unsigned short* bfW0t  = (unsigned short*)(ws + off); off += ALIGN256((size_t)HDIM * D_IN * 2);
    unsigned short* bfW1t  = (unsigned short*)(ws + off); off += ALIGN256((size_t)HDIM * HDIM * 2);
    unsigned short* bfW2t  = (unsigned short*)(ws + off); off += ALIGN256((size_t)HDIM * HDIM * 2);
    unsigned short* bfWih0 = (unsigned short*)(ws + off); off += ALIGN256((size_t)G4 * HDIM * 2);
    unsigned short* bfWih1 = (unsigned short*)(ws + off); off += ALIGN256((size_t)G4 * HDIM * 2);
    unsigned short* bfLinT = (unsigned short*)(ws + off); off += ALIGN256((size_t)D_OUT * HDIM * 2);
    float*          dinv   = (float*)(ws + off);          off += ALIGN256((size_t)N_NODES * 4);
    unsigned short* actbf  = (unsigned short*)(ws + off); off += ALIGN256((size_t)N_NODES * HDIM * 2);
    float*          bufA   = (float*)(ws + off);          off += ALIGN256((size_t)N_NODES * HDIM * 4);
    float*          bufB   = (float*)(ws + off);          off += ALIGN256((size_t)N_NODES * HDIM * 4);
    float*          whhT   = (float*)(ws + off);          off += ALIGN256((size_t)HDIM * G4 * 4);
    float*          xp     = (float*)(ws + off);          off += ALIGN256((size_t)N_NODES * G4 * 4);
    (void)ws_size; (void)in_sizes; (void)n_in; (void)out_size;

    const int TB = 256;

    // ---- weight conversion (bf16, B-transposed [N][K]) ----------------------
    k_cvt_bf16_T<<<cdiv((long long)D_IN * HDIM, TB), TB, 0, stream>>>(gcn_w[0], bfW0t, D_IN, HDIM);
    k_cvt_bf16_T<<<cdiv((long long)HDIM * HDIM, TB), TB, 0, stream>>>(gcn_w[1], bfW1t, HDIM, HDIM);
    k_cvt_bf16_T<<<cdiv((long long)HDIM * HDIM, TB), TB, 0, stream>>>(gcn_w[2], bfW2t, HDIM, HDIM);
    // wih is [4H][H] = [N][K] already
    k_cvt_bf16<<<cdiv((long long)G4 * HDIM, TB), TB, 0, stream>>>(wih0, bfWih0, G4 * HDIM);
    k_cvt_bf16<<<cdiv((long long)G4 * HDIM, TB), TB, 0, stream>>>(wih1, bfWih1, G4 * HDIM);
    k_cvt_bf16_T<<<cdiv((long long)HDIM * D_OUT, TB), TB, 0, stream>>>(lin_w, bfLinT, HDIM, D_OUT);

    // ---- degree normalization (once, reused by all 3 GCN layers) ------------
    k_deg_init <<<cdiv(N_NODES, TB), TB, 0, stream>>>(dinv, N_NODES);
    k_deg_edges<<<cdiv(N_EDGES, TB), TB, 0, stream>>>(dst, dinv, N_EDGES);
    k_dinv     <<<cdiv(N_NODES, TB), TB, 0, stream>>>(dinv, N_NODES);

    // ---- 3 GCN layers -------------------------------------------------------
    const float* act = x;           // current activation (fp32)
    int actK = D_IN;
    const unsigned short* wt[3] = { bfW0t, bfW1t, bfW2t };
    const long long scat_total = (long long)N_EDGES * (HDIM / 4);

    for (int l = 0; l < 3; ++l) {
        // A -> bf16
        k_cvt_bf16<<<cdiv((long long)N_NODES * actK, TB), TB, 0, stream>>>(act, actbf, N_NODES * actK);
        // xw = A @ W   (WMMA, LDS-staged)
        dim3 g(N_NODES / 128, HDIM / 64);
        k_gemm_bf16_wmma<<<g, TB, 0, stream>>>(actbf, wt[l], bufA,
                                               nullptr, nullptr,
                                               N_NODES, HDIM, actK, 0);
        // self-loop + bias, then edge scatter, then (maybe) relu
        k_agg_init<<<cdiv((long long)N_NODES * HDIM, TB), TB, 0, stream>>>(bufA, dinv, gcn_b[l], bufB);
        k_scatter <<<cdiv(scat_total, TB), TB, 0, stream>>>(src, dst, dinv, bufA, bufB);
        if (l < 2)
            k_relu<<<cdiv((long long)N_NODES * HDIM, TB), TB, 0, stream>>>(bufB, N_NODES * HDIM);
        act = bufB;
        actK = HDIM;
    }

    // ---- LSTM layer 0 -------------------------------------------------------
    k_cvt_bf16<<<cdiv((long long)N_NODES * HDIM, TB), TB, 0, stream>>>(bufB, actbf, N_NODES * HDIM);
    {
        dim3 g(N_NODES / 128, G4 / 64);
        k_gemm_bf16_wmma<<<g, TB, 0, stream>>>(actbf, bfWih0, xp,
                                               bih0, bhh0, N_NODES, G4, HDIM, 0);
    }
    k_transpose_f32<<<cdiv((long long)G4 * HDIM, TB), TB, 0, stream>>>(whh0, whhT, G4, HDIM);
    k_lstm_scan<<<1, HDIM, 0, stream>>>(xp, whhT, bufA, N_NODES);

    // ---- LSTM layer 1 -------------------------------------------------------
    k_cvt_bf16<<<cdiv((long long)N_NODES * HDIM, TB), TB, 0, stream>>>(bufA, actbf, N_NODES * HDIM);
    {
        dim3 g(N_NODES / 128, G4 / 64);
        k_gemm_bf16_wmma<<<g, TB, 0, stream>>>(actbf, bfWih1, xp,
                                               bih1, bhh1, N_NODES, G4, HDIM, 0);
    }
    k_transpose_f32<<<cdiv((long long)G4 * HDIM, TB), TB, 0, stream>>>(whh1, whhT, G4, HDIM);
    k_lstm_scan<<<1, HDIM, 0, stream>>>(xp, whhT, bufB, N_NODES);

    // ---- head ---------------------------------------------------------------
    k_cvt_bf16<<<cdiv((long long)N_NODES * HDIM, TB), TB, 0, stream>>>(bufB, actbf, N_NODES * HDIM);
    {
        dim3 g(N_NODES / 128, D_OUT / 64);
        k_gemm_bf16_wmma<<<g, TB, 0, stream>>>(actbf, bfLinT, out,
                                               lin_b, nullptr, N_NODES, D_OUT, HDIM, 0);
    }
}
